// PIGNN_29669634081213
// MI455X (gfx1250) — matrix operand order repaired and checked by reading, not compiled, not run on hardware
//
#include <hip/hip_runtime.h>
#include <hip/hip_bf16.h>

// -------- problem constants (match reference) --------
#define NNODES 200000
#define NEDGES 600000
#define HD     128
#define NLAY   6
#define LNEPS  1e-5f

typedef _Float16 v16h __attribute__((ext_vector_type(16)));
typedef float    v8f  __attribute__((ext_vector_type(8)));

// D = A(16x32 f16) * B(32x16 f16) + C(16x16 f32)
static __device__ inline v8f wmma16(v16h a, v16h b, v8f c) {
  return __builtin_amdgcn_wmma_f32_16x16x32_f16(
      /*neg_a=*/false, a, /*neg_b=*/false, b,
      /*c_mod=*/(short)0, c, /*reuse_a=*/false, /*reuse_b=*/false);
}

// B fragment: packed layout [frag][lane][16 f16], one 32B load per lane.
static __device__ inline v16h load_b_frag(const _Float16* __restrict__ packed, int f) {
  int lane = threadIdx.x & 31;
  return *(const v16h*)(packed + ((size_t)f * 32 + lane) * 16);
}

// A fragment from an f32 row in global memory (16-bit A layout, ISA 7.12.2):
// lane<16: row M=lane, K = kb+{0..7,16..23}; lane>=16: same row, K = kb+{8..15,24..31}
static __device__ inline v16h a_frag_f32(const float* __restrict__ row, int kb) {
  int koff = (threadIdx.x & 16) ? 8 : 0;
  const float* p = row + kb + koff;
  v16h a;
#pragma unroll
  for (int j = 0; j < 8; ++j) { a[j] = (_Float16)p[j]; a[8 + j] = (_Float16)p[16 + j]; }
  return a;
}

// A fragment from an f16 row staged in LDS
static __device__ inline v16h a_frag_lds(const _Float16* row, int kb) {
  int koff = (threadIdx.x & 16) ? 8 : 0;
  const _Float16* p = row + kb + koff;
  v16h a;
#pragma unroll
  for (int j = 0; j < 8; ++j) { a[j] = p[j]; a[8 + j] = p[16 + j]; }
  return a;
}

static __device__ inline v8f vzero8() {
  v8f z;
#pragma unroll
  for (int r = 0; r < 8; ++r) z[r] = 0.f;
  return z;
}

// -------- weight packing: f32 (K x Ho, row-major) -> f16 B-fragment layout --------
__global__ void pignn_pack(const float* __restrict__ W, _Float16* __restrict__ out,
                           int K, int Ho) {
  int t = blockIdx.x * blockDim.x + threadIdx.x;
  int total = (K / 32) * (Ho / 16) * 32;
  if (t >= total) return;
  int lane = t & 31;
  int f = t >> 5;
  int nt = f % (Ho / 16);
  int kc = f / (Ho / 16);
  int col = nt * 16 + (lane & 15);
  int kb  = kc * 32 + ((lane & 16) ? 8 : 0);
  _Float16* o = out + (size_t)t * 16;
#pragma unroll
  for (int j = 0; j < 16; ++j) o[j] = (_Float16)W[(size_t)(kb + (j < 8 ? j : j + 8)) * Ho + col];
}

// -------- encoders: in(K1 small) ->128 relu ->128 (2nd GEMM via WMMA) --------
__global__ __launch_bounds__(128) void pignn_node_encode(
    const float* __restrict__ x, const float* __restrict__ coords,
    const float* __restrict__ w1, const float* __restrict__ b1,
    const _Float16* __restrict__ w2p, const float* __restrict__ b2,
    float* __restrict__ hout, int N) {
  __shared__ float    sh_in[4][16][12];
  __shared__ _Float16 sh_hid[4][16][136];
  int lane = threadIdx.x & 31, w = threadIdx.x >> 5;
  int base = (blockIdx.x * 4 + w) * 16;
  int m = lane & 15;
  for (int idx = lane; idx < 16 * 9; idx += 32) {
    int mm = idx / 9, k = idx % 9;
    int node = base + mm; if (node >= N) node = N - 1;
    sh_in[w][mm][k] = (k < 3) ? coords[(size_t)node * 3 + k] : x[(size_t)node * 9 + k];
  }
  __syncthreads();
  for (int idx = lane; idx < 16 * 128; idx += 32) {
    int mm = idx >> 7, c = idx & 127;
    float s = b1[c];
#pragma unroll
    for (int k = 0; k < 9; ++k) s += sh_in[w][mm][k] * w1[k * 128 + c];
    sh_hid[w][mm][c] = (_Float16)fmaxf(s, 0.f);
  }
  __syncthreads();
  v8f acc[8];
#pragma unroll
  for (int nt = 0; nt < 8; ++nt) acc[nt] = vzero8();
  for (int kc = 0; kc < 4; ++kc) {
    v16h a = a_frag_lds(&sh_hid[w][m][0], kc * 32);
#pragma unroll
    for (int nt = 0; nt < 8; ++nt) acc[nt] = wmma16(a, load_b_frag(w2p, kc * 8 + nt), acc[nt]);
  }
  int col0 = lane & 15, mb = (lane & 16) ? 8 : 0;
#pragma unroll
  for (int nt = 0; nt < 8; ++nt) {
    int col = nt * 16 + col0;
    float bias = b2[col];
#pragma unroll
    for (int r = 0; r < 8; ++r) {
      int node = base + mb + r;
      if (node < N) hout[(size_t)node * HD + col] = acc[nt][r] + bias;
    }
  }
}

__global__ __launch_bounds__(128) void pignn_edge_encode(
    const float* __restrict__ ea,
    const float* __restrict__ w1, const float* __restrict__ b1,
    const _Float16* __restrict__ w2p, const float* __restrict__ b2,
    float* __restrict__ eout, int E) {
  __shared__ float    sh_in[4][16][8];
  __shared__ _Float16 sh_hid[4][16][136];
  int lane = threadIdx.x & 31, w = threadIdx.x >> 5;
  int base = (blockIdx.x * 4 + w) * 16;
  int m = lane & 15;
  for (int idx = lane; idx < 16 * 7; idx += 32) {
    int mm = idx / 7, k = idx % 7;
    int ed = base + mm; if (ed >= E) ed = E - 1;
    sh_in[w][mm][k] = ea[(size_t)ed * 7 + k];
  }
  __syncthreads();
  for (int idx = lane; idx < 16 * 128; idx += 32) {
    int mm = idx >> 7, c = idx & 127;
    float s = b1[c];
#pragma unroll
    for (int k = 0; k < 7; ++k) s += sh_in[w][mm][k] * w1[k * 128 + c];
    sh_hid[w][mm][c] = (_Float16)fmaxf(s, 0.f);
  }
  __syncthreads();
  v8f acc[8];
#pragma unroll
  for (int nt = 0; nt < 8; ++nt) acc[nt] = vzero8();
  for (int kc = 0; kc < 4; ++kc) {
    v16h a = a_frag_lds(&sh_hid[w][m][0], kc * 32);
#pragma unroll
    for (int nt = 0; nt < 8; ++nt) acc[nt] = wmma16(a, load_b_frag(w2p, kc * 8 + nt), acc[nt]);
  }
  int col0 = lane & 15, mb = (lane & 16) ? 8 : 0;
#pragma unroll
  for (int nt = 0; nt < 8; ++nt) {
    int col = nt * 16 + col0;
    float bias = b2[col];
#pragma unroll
    for (int r = 0; r < 8; ++r) {
      int ed = base + mb + r;
      if (ed < E) eout[(size_t)ed * HD + col] = acc[nt][r] + bias;
    }
  }
}

// -------- edge message: relu([h_s|h_d|e] @ W1 + b1) @ W2 + b2 -> atomic agg[dst] --------
__global__ __launch_bounds__(128) void pignn_edge_msg(
    const float* __restrict__ h, const float* __restrict__ e,
    const int* __restrict__ src, const int* __restrict__ dst,
    const _Float16* __restrict__ w1p, const float* __restrict__ b1,
    const _Float16* __restrict__ w2p, const float* __restrict__ b2,
    float* __restrict__ agg, int E) {
  __shared__ _Float16 sh_hid[4][16][136];
  __shared__ int      sh_dst[4][16];
  int lane = threadIdx.x & 31, w = threadIdx.x >> 5;
  int base = (blockIdx.x * 4 + w) * 16;
  int m = lane & 15;
  int erow = base + m; if (erow >= E) erow = E - 1;
  int my_src = src[erow], my_dst = dst[erow];
  if (lane < 16) sh_dst[w][m] = my_dst;
  const float* srow = h + (size_t)my_src * HD;
  const float* drow = h + (size_t)my_dst * HD;
  const float* arow = e + (size_t)erow * HD;

  v8f acc[8];
#pragma unroll
  for (int nt = 0; nt < 8; ++nt) acc[nt] = vzero8();
  for (int kc = 0; kc < 12; ++kc) {           // K = 384
    const float* rp; int kb;
    if (kc < 4)      { rp = srow; kb = kc * 32; }
    else if (kc < 8) { rp = drow; kb = (kc - 4) * 32; }
    else             { rp = arow; kb = (kc - 8) * 32; }
    v16h a = a_frag_f32(rp, kb);
#pragma unroll
    for (int nt = 0; nt < 8; ++nt) acc[nt] = wmma16(a, load_b_frag(w1p, kc * 8 + nt), acc[nt]);
  }
  int col0 = lane & 15, mb = (lane & 16) ? 8 : 0;
#pragma unroll
  for (int nt = 0; nt < 8; ++nt) {
    int col = nt * 16 + col0;
    float bias = b1[col];
#pragma unroll
    for (int r = 0; r < 8; ++r)
      sh_hid[w][mb + r][col] = (_Float16)fmaxf(acc[nt][r] + bias, 0.f);
  }
  __syncthreads();
  v8f acc2[8];
#pragma unroll
  for (int nt = 0; nt < 8; ++nt) acc2[nt] = vzero8();
  for (int kc = 0; kc < 4; ++kc) {            // K = 128
    v16h a = a_frag_lds(&sh_hid[w][m][0], kc * 32);
#pragma unroll
    for (int nt = 0; nt < 8; ++nt) acc2[nt] = wmma16(a, load_b_frag(w2p, kc * 8 + nt), acc2[nt]);
  }
#pragma unroll
  for (int nt = 0; nt < 8; ++nt) {
    int col = nt * 16 + col0;
    float bias = b2[col];
#pragma unroll
    for (int r = 0; r < 8; ++r) {
      int rr = mb + r;
      if (base + rr < E) {
        int d = sh_dst[w][rr];
        atomicAdd(&agg[(size_t)d * HD + col], acc2[nt][r] + bias);
      }
    }
  }
}

// -------- node update: relu([h|agg]@W1+b1)@W2+b2, residual + LayerNorm --------
__global__ __launch_bounds__(128) void pignn_node_update(
    const float* __restrict__ h, const float* __restrict__ agg,
    const _Float16* __restrict__ w1p, const float* __restrict__ b1,
    const _Float16* __restrict__ w2p, const float* __restrict__ b2,
    const float* __restrict__ lng, const float* __restrict__ lnb,
    float* __restrict__ hout, int N) {
  __shared__ _Float16 sh_hid[4][16][136];
  __shared__ float    sh_pre[4][16][132];
  int lane = threadIdx.x & 31, w = threadIdx.x >> 5;
  int base = (blockIdx.x * 4 + w) * 16;
  int m = lane & 15;
  int node = base + m; if (node >= N) node = N - 1;
  const float* hrow = h + (size_t)node * HD;
  const float* arow = agg + (size_t)node * HD;

  v8f acc[8];
#pragma unroll
  for (int nt = 0; nt < 8; ++nt) acc[nt] = vzero8();
  for (int kc = 0; kc < 8; ++kc) {            // K = 256
    const float* rp = (kc < 4) ? hrow : arow;
    int kb = (kc & 3) * 32;
    v16h a = a_frag_f32(rp, kb);
#pragma unroll
    for (int nt = 0; nt < 8; ++nt) acc[nt] = wmma16(a, load_b_frag(w1p, kc * 8 + nt), acc[nt]);
  }
  int col0 = lane & 15, mb = (lane & 16) ? 8 : 0;
#pragma unroll
  for (int nt = 0; nt < 8; ++nt) {
    int col = nt * 16 + col0;
    float bias = b1[col];
#pragma unroll
    for (int r = 0; r < 8; ++r)
      sh_hid[w][mb + r][col] = (_Float16)fmaxf(acc[nt][r] + bias, 0.f);
  }
  __syncthreads();
  v8f acc2[8];
#pragma unroll
  for (int nt = 0; nt < 8; ++nt) acc2[nt] = vzero8();
  for (int kc = 0; kc < 4; ++kc) {
    v16h a = a_frag_lds(&sh_hid[w][m][0], kc * 32);
#pragma unroll
    for (int nt = 0; nt < 8; ++nt) acc2[nt] = wmma16(a, load_b_frag(w2p, kc * 8 + nt), acc2[nt]);
  }
#pragma unroll
  for (int nt = 0; nt < 8; ++nt) {
    int col = nt * 16 + col0;
    float bias = b2[col];
#pragma unroll
    for (int r = 0; r < 8; ++r) {
      int rr = mb + r;
      int nn = base + rr; if (nn >= N) nn = N - 1;
      sh_pre[w][rr][col] = acc2[nt][r] + bias + h[(size_t)nn * HD + col];
    }
  }
  __syncthreads();
  // LayerNorm: lane handles one half-row (lane<16 -> cols 0..63, else 64..127)
  int c0 = (lane & 16) ? 64 : 0;
  const float* prow = &sh_pre[w][m][0];
  float s = 0.f, s2 = 0.f;
#pragma unroll 8
  for (int c = 0; c < 64; ++c) { float v = prow[c0 + c]; s += v; s2 += v * v; }
  s  += __shfl_xor(s, 16, 32);
  s2 += __shfl_xor(s2, 16, 32);
  float mean = s * (1.f / 128.f);
  float var  = s2 * (1.f / 128.f) - mean * mean;
  float rstd = rsqrtf(var + LNEPS);
  if (base + m < N) {
    float* orow = hout + (size_t)(base + m) * HD;
#pragma unroll 8
    for (int c = 0; c < 64; ++c) {
      int cc = c0 + c;
      orow[cc] = (prow[cc] - mean) * rstd * lng[cc] + lnb[cc];
    }
  }
}

// -------- decoder: relu(h@W1+b1) -> relu(@W2+b2, Ho=64) -> @W3+b3 (3) + BC mask --------
__global__ __launch_bounds__(128) void pignn_decode(
    const float* __restrict__ h,
    const _Float16* __restrict__ w1p, const float* __restrict__ b1,
    const _Float16* __restrict__ w2p, const float* __restrict__ b2,
    const float* __restrict__ w3, const float* __restrict__ b3,
    const float* __restrict__ bc_disp, const float* __restrict__ bc_rot,
    float* __restrict__ out, int N) {
  __shared__ _Float16 sh_hid[4][16][136];
  __shared__ float    sh_h2[4][16][68];
  int lane = threadIdx.x & 31, w = threadIdx.x >> 5;
  int base = (blockIdx.x * 4 + w) * 16;
  int m = lane & 15;
  int node = base + m; if (node >= N) node = N - 1;
  const float* hrow = h + (size_t)node * HD;

  v8f acc[8];
#pragma unroll
  for (int nt = 0; nt < 8; ++nt) acc[nt] = vzero8();
  for (int kc = 0; kc < 4; ++kc) {
    v16h a = a_frag_f32(hrow, kc * 32);
#pragma unroll
    for (int nt = 0; nt < 8; ++nt) acc[nt] = wmma16(a, load_b_frag(w1p, kc * 8 + nt), acc[nt]);
  }
  int col0 = lane & 15, mb = (lane & 16) ? 8 : 0;
#pragma unroll
  for (int nt = 0; nt < 8; ++nt) {
    int col = nt * 16 + col0;
    float bias = b1[col];
#pragma unroll
    for (int r = 0; r < 8; ++r)
      sh_hid[w][mb + r][col] = (_Float16)fmaxf(acc[nt][r] + bias, 0.f);
  }
  __syncthreads();
  v8f acc2[4];
#pragma unroll
  for (int nt = 0; nt < 4; ++nt) acc2[nt] = vzero8();
  for (int kc = 0; kc < 4; ++kc) {
    v16h a = a_frag_lds(&sh_hid[w][m][0], kc * 32);
#pragma unroll
    for (int nt = 0; nt < 4; ++nt) acc2[nt] = wmma16(a, load_b_frag(w2p, kc * 4 + nt), acc2[nt]);
  }
#pragma unroll
  for (int nt = 0; nt < 4; ++nt) {
    int col = nt * 16 + col0;
    float bias = b2[col];
#pragma unroll
    for (int r = 0; r < 8; ++r)
      sh_h2[w][mb + r][col] = fmaxf(acc2[nt][r] + bias, 0.f);
  }
  __syncthreads();
  if (lane < 16 && base + m < N) {
    int nn = base + m;
    float p0 = b3[0], p1 = b3[1], p2 = b3[2];
#pragma unroll 8
    for (int k = 0; k < 64; ++k) {
      float v = sh_h2[w][m][k];
      p0 += v * w3[k * 3 + 0];
      p1 += v * w3[k * 3 + 1];
      p2 += v * w3[k * 3 + 2];
    }
    float md = 1.f - bc_disp[nn];
    float mr = 1.f - bc_rot[nn];
    out[(size_t)nn * 3 + 0] = p0 * md;
    out[(size_t)nn * 3 + 1] = p1 * md;
    out[(size_t)nn * 3 + 2] = p2 * mr;
  }
}

// -------------------------- host orchestration --------------------------
extern "C" void kernel_launch(void* const* d_in, const int* in_sizes, int n_in,
                              void* d_out, int out_size, void* d_ws, size_t ws_size,
                              hipStream_t stream) {
  (void)in_sizes; (void)n_in; (void)out_size; (void)ws_size;
  const float* x         = (const float*)d_in[0];
  const float* coords    = (const float*)d_in[1];
  const float* edge_attr = (const float*)d_in[2];
  const int*   eidx      = (const int*)d_in[3];
  const float* bc_disp   = (const float*)d_in[4];
  const float* bc_rot    = (const float*)d_in[5];
  const float* ne_w1 = (const float*)d_in[6];  const float* ne_b1 = (const float*)d_in[7];
  const float* ne_w2 = (const float*)d_in[8];  const float* ne_b2 = (const float*)d_in[9];
  const float* ee_w1 = (const float*)d_in[10]; const float* ee_b1 = (const float*)d_in[11];
  const float* ee_w2 = (const float*)d_in[12]; const float* ee_b2 = (const float*)d_in[13];
  const float* em_w1 = (const float*)d_in[14]; const float* em_b1 = (const float*)d_in[15];
  const float* em_w2 = (const float*)d_in[16]; const float* em_b2 = (const float*)d_in[17];
  const float* nm_w1 = (const float*)d_in[18]; const float* nm_b1 = (const float*)d_in[19];
  const float* nm_w2 = (const float*)d_in[20]; const float* nm_b2 = (const float*)d_in[21];
  const float* ln_g  = (const float*)d_in[22]; const float* ln_b  = (const float*)d_in[23];
  const float* dc_w1 = (const float*)d_in[24]; const float* dc_b1 = (const float*)d_in[25];
  const float* dc_w2 = (const float*)d_in[26]; const float* dc_b2 = (const float*)d_in[27];
  const float* dc_w3 = (const float*)d_in[28]; const float* dc_b3 = (const float*)d_in[29];

  char* ws = (char*)d_ws;
  size_t off = 0;
  auto carve = [&](size_t bytes) -> char* {
    char* p = ws + off; off += (bytes + 255) & ~(size_t)255; return p;
  };
  float* hbuf = (float*)carve((size_t)NNODES * HD * 4);
  float* agg  = (float*)carve((size_t)NNODES * HD * 4);
  float* ebuf = (float*)carve((size_t)NEDGES * HD * 4);
  _Float16* pk_ne_w2 = (_Float16*)carve((size_t)128 * 128 * 2);
  _Float16* pk_ee_w2 = (_Float16*)carve((size_t)128 * 128 * 2);
  _Float16* pk_em_w1 = (_Float16*)carve((size_t)NLAY * 384 * 128 * 2);
  _Float16* pk_em_w2 = (_Float16*)carve((size_t)NLAY * 128 * 128 * 2);
  _Float16* pk_nm_w1 = (_Float16*)carve((size_t)NLAY * 256 * 128 * 2);
  _Float16* pk_nm_w2 = (_Float16*)carve((size_t)NLAY * 128 * 128 * 2);
  _Float16* pk_dc_w1 = (_Float16*)carve((size_t)128 * 128 * 2);
  _Float16* pk_dc_w2 = (_Float16*)carve((size_t)128 * 64 * 2);

  auto pack = [&](const float* W, _Float16* dst, int K, int Ho) {
    int total = (K / 32) * (Ho / 16) * 32;
    pignn_pack<<<(total + 255) / 256, 256, 0, stream>>>(W, dst, K, Ho);
  };
  pack(ne_w2, pk_ne_w2, 128, 128);
  pack(ee_w2, pk_ee_w2, 128, 128);
  for (int l = 0; l < NLAY; ++l) {
    pack(em_w1 + (size_t)l * 384 * 128, pk_em_w1 + (size_t)l * 384 * 128, 384, 128);
    pack(em_w2 + (size_t)l * 128 * 128, pk_em_w2 + (size_t)l * 128 * 128, 128, 128);
    pack(nm_w1 + (size_t)l * 256 * 128, pk_nm_w1 + (size_t)l * 256 * 128, 256, 128);
    pack(nm_w2 + (size_t)l * 128 * 128, pk_nm_w2 + (size_t)l * 128 * 128, 128, 128);
  }
  pack(dc_w1, pk_dc_w1, 128, 128);
  pack(dc_w2, pk_dc_w2, 128, 64);

  int nblk = (NNODES + 63) / 64;
  int eblk = (NEDGES + 63) / 64;

  pignn_node_encode<<<nblk, 128, 0, stream>>>(x, coords, ne_w1, ne_b1, pk_ne_w2, ne_b2,
                                              hbuf, NNODES);
  pignn_edge_encode<<<eblk, 128, 0, stream>>>(edge_attr, ee_w1, ee_b1, pk_ee_w2, ee_b2,
                                              ebuf, NEDGES);

  const int* src = eidx;
  const int* dst = eidx + NEDGES;
  for (int l = 0; l < NLAY; ++l) {
    hipMemsetAsync(agg, 0, (size_t)NNODES * HD * 4, stream);
    pignn_edge_msg<<<eblk, 128, 0, stream>>>(
        hbuf, ebuf, src, dst,
        pk_em_w1 + (size_t)l * 384 * 128, em_b1 + (size_t)l * HD,
        pk_em_w2 + (size_t)l * 128 * 128, em_b2 + (size_t)l * HD,
        agg, NEDGES);
    pignn_node_update<<<nblk, 128, 0, stream>>>(
        hbuf, agg,
        pk_nm_w1 + (size_t)l * 256 * 128, nm_b1 + (size_t)l * HD,
        pk_nm_w2 + (size_t)l * 128 * 128, nm_b2 + (size_t)l * HD,
        ln_g + (size_t)l * HD, ln_b + (size_t)l * HD,
        hbuf, NNODES);
  }

  pignn_decode<<<nblk, 128, 0, stream>>>(hbuf, pk_dc_w1, dc_b1, pk_dc_w2, dc_b2,
                                         dc_w3, dc_b3, bc_disp, bc_rot,
                                         (float*)d_out, NNODES);
}
